// RIMAEBackbone_46067819216911
// MI455X (gfx1250) — matrix operand order repaired and checked by reading, not compiled, not run on hardware
//
#include <hip/hip_runtime.h>

#define DEV __device__ __forceinline__

typedef __bf16 bf16;
typedef __attribute__((ext_vector_type(16))) __bf16 v16bf;
typedef __attribute__((ext_vector_type(8)))  __bf16 v8bf;
typedef __attribute__((ext_vector_type(8)))  float  v8f;

namespace {
constexpr int cB = 4, cN = 8192, cG = 512, cS = 32, cE = 384;
constexpr int cHEADS = 6, cHD = 64, cDEPTH = 4, cHM = 1536;
constexpr int BG = cB * cG;                 // 2048 groups / tokens
constexpr float cEPS = 1e-5f;
constexpr float cBNI = 0.9999950000374998f; // 1/sqrt(1+eps)
constexpr unsigned ENC_LDS_BYTES = 117760;  // F2(32K)+F3(64K)+F1(16K)+GF(2K)+NE
constexpr unsigned ATT_LDS_BYTES = 131072;  // 64x512 f32 scores (bf16 P overlaid)
}

// ---------------------------------------------------------------- WMMA helpers
DEV v8f wmma_bf16(v16bf a, v16bf b, v8f c) {
  return __builtin_amdgcn_wmma_f32_16x16x32_bf16(false, a, false, b, (short)0, c,
                                                 false, false);
}
DEV v8f vzero8() { v8f z = {0.f,0.f,0.f,0.f,0.f,0.f,0.f,0.f}; return z; }

// A-matrix (16x32 bf16, MxK): lane l holds row m0+(l&15); elems 0..15 = K (l>>4)*16..
// (16 contiguous K, loaded as two 16B chunks)
DEV v16bf load_a16(const bf16* base, int ld, int m0, int k0, int lane) {
  const bf16* p = base + (size_t)(m0 + (lane & 15)) * ld + k0 + ((lane >> 4) << 4);
  union { v16bf v; v8bf h[2]; } u;
  u.h[0] = *(const v8bf*)(p);
  u.h[1] = *(const v8bf*)(p + 8);
  return u.v;
}
// B-matrix (32x16 bf16, KxN) from row-major [N][K] storage: lane l holds column
// n0+(l&15); 16 contiguous K at k0+(l>>4)*16  (one 32B chunk)
DEV v16bf load_b16(const bf16* base, int ld, int n0, int k0, int lane) {
  const bf16* p = base + (size_t)(n0 + (lane & 15)) * ld + k0 + ((lane >> 4) << 4);
  return *(const v16bf*)p;
}
// C/D (16x16 f32): lane l: n = l&15, VGPR i -> m = i + 8*(l>>4)

DEV float gelu_exact(float x) { return 0.5f * x * (1.0f + erff(x * 0.70710678118654752f)); }

// ---------------------------------------------------------------- FPS (serial)
__global__ __launch_bounds__(256) void k_fps(const float* __restrict__ xyz,
                                             int* __restrict__ fps_idx) {
  const int b = blockIdx.x, tid = threadIdx.x;
  const float* P = xyz + (size_t)b * cN * 3;
  float dmin[32];
#pragma unroll
  for (int i = 0; i < 32; ++i) dmin[i] = 1e10f;
  __shared__ float sv[256];
  __shared__ int   si[256];
  int far = 0;
  for (int g = 0; g < cG; ++g) {
    if (tid == 0) fps_idx[b * cG + g] = far;
    const float cx = P[far * 3 + 0], cy = P[far * 3 + 1], cz = P[far * 3 + 2];
    float bestv = -1.f; int besti = 0;
#pragma unroll 4
    for (int i = 0; i < 32; ++i) {
      const int p = tid + (i << 8);
      const float dx = P[p * 3 + 0] - cx, dy = P[p * 3 + 1] - cy, dz = P[p * 3 + 2] - cz;
      const float d = dx * dx + dy * dy + dz * dz;
      const float m = fminf(dmin[i], d);
      dmin[i] = m;
      if (m > bestv) { bestv = m; besti = p; }
    }
    sv[tid] = bestv; si[tid] = besti;
    __syncthreads();
    for (int off = 128; off > 0; off >>= 1) {
      if (tid < off && sv[tid + off] > sv[tid]) { sv[tid] = sv[tid + off]; si[tid] = si[tid + off]; }
      __syncthreads();
    }
    far = si[0];
    __syncthreads();
  }
}

// ---------------------------------------------------------------- KNN top-32
__global__ __launch_bounds__(256) void k_knn(const float* __restrict__ xyz,
                                             const int* __restrict__ fps_idx,
                                             int* __restrict__ group_idx,
                                             float* __restrict__ center) {
  const int bg = blockIdx.x, b = bg / cG, tid = threadIdx.x;
  const float* P = xyz + (size_t)b * cN * 3;
  const int ci = fps_idx[bg];
  const float cx = P[ci * 3 + 0], cy = P[ci * 3 + 1], cz = P[ci * 3 + 2];
  if (tid == 0) { center[bg * 3 + 0] = cx; center[bg * 3 + 1] = cy; center[bg * 3 + 2] = cz; }
  __shared__ float d2[cN];
  __shared__ float sv[256];
  __shared__ int   si[256];
  for (int p = tid; p < cN; p += 256) {
    const float dx = P[p * 3 + 0] - cx, dy = P[p * 3 + 1] - cy, dz = P[p * 3 + 2] - cz;
    d2[p] = dx * dx + dy * dy + dz * dz;
  }
  __syncthreads();
  for (int s = 0; s < cS; ++s) {
    float bestv = 1e30f; int besti = 0;
    for (int p = tid; p < cN; p += 256) {
      const float d = d2[p];
      if (d < bestv) { bestv = d; besti = p; }
    }
    sv[tid] = bestv; si[tid] = besti;
    __syncthreads();
    for (int off = 128; off > 0; off >>= 1) {
      if (tid < off && sv[tid + off] < sv[tid]) { sv[tid] = sv[tid + off]; si[tid] = si[tid + off]; }
      __syncthreads();
    }
    if (tid == 0) { group_idx[bg * cS + s] = si[0]; d2[si[0]] = 1e30f; }
    __syncthreads();
  }
}

// ---------------------------------------------------------------- greedy order
__global__ __launch_bounds__(512) void k_order(const float* __restrict__ center,
                                               int* __restrict__ order) {
  const int b = blockIdx.x, tid = threadIdx.x; // 512 threads, one per center
  __shared__ float cs[cG * 3];
  __shared__ unsigned char vis[cG];
  __shared__ float sv[512];
  __shared__ int   si[512];
  cs[tid * 3 + 0] = center[((size_t)b * cG + tid) * 3 + 0];
  cs[tid * 3 + 1] = center[((size_t)b * cG + tid) * 3 + 1];
  cs[tid * 3 + 2] = center[((size_t)b * cG + tid) * 3 + 2];
  vis[tid] = (tid == 0);
  if (tid == 0) order[b * cG] = 0;
  __syncthreads();
  int last = 0;
  for (int g = 1; g < cG; ++g) {
    float v = 1e30f;
    if (!vis[tid]) {
      const float dx = cs[tid * 3 + 0] - cs[last * 3 + 0];
      const float dy = cs[tid * 3 + 1] - cs[last * 3 + 1];
      const float dz = cs[tid * 3 + 2] - cs[last * 3 + 2];
      v = dx * dx + dy * dy + dz * dz;
    }
    sv[tid] = v; si[tid] = tid;
    __syncthreads();
    for (int off = 256; off > 0; off >>= 1) {
      if (tid < off && sv[tid + off] < sv[tid]) { sv[tid] = sv[tid + off]; si[tid] = si[tid + off]; }
      __syncthreads();
    }
    last = si[0];
    if (tid == 0) { vis[last] = 1; order[b * cG + g] = last; }
    __syncthreads();
  }
}

// ---------------------------------------------------------------- gather neigh
__global__ void k_gather(const float* __restrict__ xyz, const float* __restrict__ center,
                         const int* __restrict__ group_idx, const int* __restrict__ order,
                         float* __restrict__ neigh, float* __restrict__ cord) {
  const int b = blockIdx.x / cG, go = blockIdx.x % cG, tid = threadIdx.x; // 96 threads
  const int gs = order[b * cG + go];
  const int bgs = b * cG + gs;
  const float* P = xyz + (size_t)b * cN * 3;
  if (tid < 3) cord[((size_t)b * cG + go) * 3 + tid] = center[bgs * 3 + tid];
  const int s = tid / 3, j = tid % 3;
  if (s < cS) {
    const int p = group_idx[bgs * cS + s];
    neigh[(((size_t)b * cG + go) * cS + s) * 3 + j] = P[p * 3 + j] - center[bgs * 3 + j];
  }
}

// ---------------------------------------------------------------- f32 -> bf16
__global__ void k_f2bf(const float* __restrict__ s, bf16* __restrict__ d, int n) {
  const int i = blockIdx.x * 256 + threadIdx.x;
  if (i < n) d[i] = (bf16)s[i];
}

// ---------------------------------------------------------------- fused encoder
// One workgroup per group; activations [s][chan] bf16 in LDS; weights bf16 from L2.
__global__ __launch_bounds__(256) void k_enc(
    const float* __restrict__ neigh, const float* __restrict__ cord,
    const float* __restrict__ w1, const float* __restrict__ b1,
    const float* __restrict__ g1, const float* __restrict__ bt1,
    const float* __restrict__ b2, const float* __restrict__ g2, const float* __restrict__ bt2,
    const float* __restrict__ b3,
    const float* __restrict__ b4, const float* __restrict__ g4, const float* __restrict__ bt4,
    const float* __restrict__ b5,
    const bf16* __restrict__ w2b, const bf16* __restrict__ w3b,
    const bf16* __restrict__ w4b, const bf16* __restrict__ w5b,
    float* __restrict__ X) {
  extern __shared__ char lds[];
  bf16*  F2 = (bf16*)(lds);                  // [32][512]
  bf16*  F3 = (bf16*)(lds + 32768);          // [32][1024]
  float* F5 = (float*)(lds + 32768);         // [32][384] alias (after F3 dead)
  bf16*  F1 = (bf16*)(lds + 98304);          // [32][256] f1, reused as f4 chunk
  bf16*  GF = (bf16*)(lds + 114688);         // [1024] global feature
  float* NE = (float*)(lds + 116736);        // [32][3]
  const int bg = blockIdx.x, tid = threadIdx.x;
  const int w = tid >> 5, lane = tid & 31;

  if (tid < 96) NE[tid] = neigh[(size_t)bg * 96 + tid];
  __syncthreads();

  // ---- f1 = relu(bn1(W1 (3->256) x))  [VALU: K=3 too small for WMMA]
  for (int e = tid; e < 256 * cS; e += 256) {
    const int s = e & 31, c = e >> 5;
    float v = w1[c * 3 + 0] * NE[s * 3 + 0] + w1[c * 3 + 1] * NE[s * 3 + 1] +
              w1[c * 3 + 2] * NE[s * 3 + 2] + b1[c];
    v = v * g1[c] * cBNI + bt1[c];
    F1[s * 256 + c] = (bf16)fmaxf(v, 0.f);
  }
  __syncthreads();

  // ---- f2 = relu(bn2(W2 (256->512) f1)) : 32 Mtiles x 2 Ntiles
  for (int t = w; t < 64; t += 8) {
    const int mt = t >> 1, nt = t & 1;
    v8f acc = vzero8();
    for (int k = 0; k < 256; k += 32)
      acc = wmma_bf16(load_a16(w2b, 256, mt * 16, k, lane),
                      load_b16(F1, 256, nt * 16, k, lane), acc);
    const int sb = nt * 16 + (lane & 15), cb = mt * 16 + ((lane >> 4) << 3);
    v8bf o;
#pragma unroll
    for (int i = 0; i < 8; ++i) {
      const int c = cb + i;
      float v = acc[i] + b2[c];
      v = v * g2[c] * cBNI + bt2[c];
      o[i] = (bf16)fmaxf(v, 0.f);
    }
    *(v8bf*)(F2 + sb * 512 + cb) = o;
  }
  __syncthreads();

  // ---- f3 = W3 (512->1024) f2 + b3 : 64 Mtiles x 2 Ntiles
  for (int t = w; t < 128; t += 8) {
    const int mt = t >> 1, nt = t & 1;
    v8f acc = vzero8();
    for (int k = 0; k < 512; k += 32)
      acc = wmma_bf16(load_a16(w3b, 512, mt * 16, k, lane),
                      load_b16(F2, 512, nt * 16, k, lane), acc);
    const int sb = nt * 16 + (lane & 15), cb = mt * 16 + ((lane >> 4) << 3);
    v8bf o;
#pragma unroll
    for (int i = 0; i < 8; ++i) o[i] = (bf16)(acc[i] + b3[cb + i]);
    *(v8bf*)(F3 + sb * 1024 + cb) = o;
  }
  __syncthreads();

  // ---- gf = max_s f3  (broadcast half of concat; kept as a vector)
  for (int c = tid; c < 1024; c += 256) {
    float m = -1e30f;
    for (int s = 0; s < cS; ++s) m = fmaxf(m, (float)F3[s * 1024 + c]);
    GF[c] = (bf16)m;
  }
  __syncthreads();

  // ---- f4 = relu(bn4(W4 (2048->2048) [gf;f3])), f5 = W5 (2048->384) f4 + b5
  // f4 computed in 8 chunks of 256 channels; f5 accumulated in registers.
  v8f facc[6];
#pragma unroll
  for (int j = 0; j < 6; ++j) facc[j] = vzero8();
  for (int ch = 0; ch < 8; ++ch) {
    for (int t = w; t < 32; t += 8) {                 // f4 chunk: 16 Mtiles x 2
      const int mt = t >> 1, nt = t & 1;
      v8f acc = vzero8();
      for (int k = 0; k < 2048; k += 32) {
        const v16bf a = load_a16(w4b, 2048, ch * 256 + mt * 16, k, lane);
        v16bf bm;
        if (k < 1024) {                               // gf broadcast over s
          bm = *(const v16bf*)(GF + k + ((lane >> 4) << 4));
        } else {
          bm = load_b16(F3, 1024, nt * 16, k - 1024, lane);
        }
        acc = wmma_bf16(a, bm, acc);
      }
      const int sb = nt * 16 + (lane & 15), cb = mt * 16 + ((lane >> 4) << 3);
      v8bf o;
#pragma unroll
      for (int i = 0; i < 8; ++i) {
        const int c = ch * 256 + cb + i;
        float v = acc[i] + b4[c];
        v = v * g4[c] * cBNI + bt4[c];
        o[i] = (bf16)fmaxf(v, 0.f);
      }
      *(v8bf*)(F1 + sb * 256 + cb) = o;
    }
    __syncthreads();
    for (int j = 0; j < 6; ++j) {                     // f5 += W5[:,chunk] * f4chunk
      const int t5 = w + 8 * j, mt = t5 >> 1, nt = t5 & 1;
      v8f acc = facc[j];
      for (int k = 0; k < 256; k += 32)
        acc = wmma_bf16(load_a16(w5b, 2048, mt * 16, ch * 256 + k, lane),
                        load_b16(F1, 256, nt * 16, k, lane), acc);
      facc[j] = acc;
    }
    __syncthreads();
  }
  // spill f5 to LDS (F3 region is dead now)
  for (int j = 0; j < 6; ++j) {
    const int t5 = w + 8 * j, mt = t5 >> 1, nt = t5 & 1;
    const int sb = nt * 16 + (lane & 15), cb = mt * 16 + ((lane >> 4) << 3);
#pragma unroll
    for (int i = 0; i < 8; ++i) F5[sb * 384 + cb + i] = facc[j][i] + b5[cb + i];
  }
  __syncthreads();

  // ---- token = max_s f5 ; add sincos positional embedding; write residual X
  const float c0 = cord[(size_t)bg * 3 + 0], c1 = cord[(size_t)bg * 3 + 1],
              c2 = cord[(size_t)bg * 3 + 2];
  for (int c = tid; c < cE; c += 256) {
    float m = -1e30f;
    for (int s = 0; s < cS; ++s) m = fmaxf(m, F5[s * 384 + c]);
    const int jdim = c / 128, r = (c & 127) >> 1;
    const float ctr = (jdim == 0) ? c0 : (jdim == 1 ? c1 : c2);
    const float dimv = powf(10000.f, (float)r * (1.0f / 64.0f));
    const float ang = ctr * 6.283185307179586f / dimv;
    const float pos = (c & 1) ? cosf(ang) : sinf(ang);
    X[(size_t)bg * cE + c] = m + pos;
  }
}

// ---------------------------------------------------------------- layernorm
__global__ __launch_bounds__(128) void k_ln(const float* __restrict__ X,
                                            const float* __restrict__ g,
                                            const float* __restrict__ b,
                                            bf16* __restrict__ Y) {
  const int t = blockIdx.x, tid = threadIdx.x;
  const float* x = X + (size_t)t * cE;
  __shared__ float red[128];
  float s = 0.f;
  for (int i = tid; i < cE; i += 128) s += x[i];
  red[tid] = s; __syncthreads();
  for (int o = 64; o > 0; o >>= 1) { if (tid < o) red[tid] += red[tid + o]; __syncthreads(); }
  const float mean = red[0] * (1.0f / cE);
  __syncthreads();
  float v = 0.f;
  for (int i = tid; i < cE; i += 128) { const float d = x[i] - mean; v += d * d; }
  red[tid] = v; __syncthreads();
  for (int o = 64; o > 0; o >>= 1) { if (tid < o) red[tid] += red[tid + o]; __syncthreads(); }
  const float rs = rsqrtf(red[0] * (1.0f / cE) + cEPS);
  for (int i = tid; i < cE; i += 128)
    Y[(size_t)t * cE + i] = (bf16)((x[i] - mean) * rs * g[i] + b[i]);
}

// ---------------------------------------------------------------- generic GEMM
// C[tok, col] = A[2048xK] * W[NoutxK]^T + bias.  One 16x16 tile per wave.
// MODE 1: residual add into f32 XR. MODE 2: exact-GELU, bf16 OUT.
// MODE 3: qkv routing — q,k to OUT[2048][1152]; v transposed into VT[B][384][512].
template <int MODE>
__global__ __launch_bounds__(256) void k_gemm(const bf16* __restrict__ A,
                                              const bf16* __restrict__ W,
                                              const float* __restrict__ bias,
                                              float* __restrict__ XR,
                                              bf16* __restrict__ OUT,
                                              bf16* __restrict__ VT,
                                              int K, int Ntiles) {
  const int w = threadIdx.x >> 5, lane = threadIdx.x & 31;
  const int id = blockIdx.x * 8 + w;
  const int mt = id / Ntiles, nt = id % Ntiles;
  v8f acc = vzero8();
  for (int k = 0; k < K; k += 32) {
    __builtin_prefetch(W + (size_t)(nt * 16 + (lane & 15)) * K + k + 128, 0, 0);
    acc = wmma_bf16(load_a16(A, K, mt * 16, k, lane),
                    load_b16(W, K, nt * 16, k, lane), acc);
  }
  const int Nout = Ntiles * 16;
  const int col = nt * 16 + (lane & 15);
  const float bc = bias[col];
#pragma unroll
  for (int i = 0; i < 8; ++i) {
    const int tok = mt * 16 + ((lane >> 4) << 3) + i;
    const float v = acc[i] + bc;
    if (MODE == 1) {
      XR[(size_t)tok * Nout + col] += v;
    } else if (MODE == 2) {
      OUT[(size_t)tok * Nout + col] = (bf16)gelu_exact(v);
    } else { // MODE == 3 (qkv)
      if (col < 768) OUT[(size_t)tok * 1152 + col] = (bf16)v;
      else VT[((size_t)(tok >> 9) * cE + (col - 768)) * cG + (tok & 511)] = (bf16)v;
    }
  }
}

// ---------------------------------------------------------------- attention
// One block per (batch, head, 64-query block). S=QK^T (WMMA), softmax (LDS f32,
// bf16 P written behind the read pointer), O=P*V (WMMA vs transposed V).
__global__ __launch_bounds__(256) void k_attn(const bf16* __restrict__ QKV,
                                              const bf16* __restrict__ VT,
                                              bf16* __restrict__ OBF) {
  extern __shared__ char lds[];
  float* SC = (float*)lds;   // [64][512] scores
  bf16*  PB = (bf16*)lds;    // [64] rows at stride 1024, overlaid per-row
  const int tid = threadIdx.x, w = tid >> 5, lane = tid & 31;
  const int b = blockIdx.x / (cHEADS * 8);
  const int h = (blockIdx.x / 8) % cHEADS;
  const int qb = blockIdx.x & 7;
  const int qrow0 = b * cG + qb * 64;

  for (int t = w; t < 128; t += 8) {                  // 4 Mtiles x 32 Ntiles
    const int mt = t >> 5, nt = t & 31;
    v8f acc = vzero8();
    for (int k = 0; k < cHD; k += 32)
      acc = wmma_bf16(load_a16(QKV, 1152, qrow0 + mt * 16, h * cHD + k, lane),
                      load_b16(QKV + 384, 1152, b * cG + nt * 16, h * cHD + k, lane), acc);
    const int n = nt * 16 + (lane & 15);
#pragma unroll
    for (int i = 0; i < 8; ++i) {
      const int m = mt * 16 + ((lane >> 4) << 3) + i;
      SC[m * 512 + n] = acc[i] * 0.125f;              // 1/sqrt(64)
    }
  }
  __syncthreads();
  if (tid < 64) {                                     // softmax per query row
    float mx = -1e30f;
    for (int c = 0; c < cG; ++c) mx = fmaxf(mx, SC[tid * 512 + c]);
    float sum = 0.f;
    for (int c = 0; c < cG; ++c) { const float e = expf(SC[tid * 512 + c] - mx); SC[tid * 512 + c] = e; sum += e; }
    const float inv = 1.0f / sum;
    for (int c = 0; c < cG; ++c) PB[tid * 1024 + c] = (bf16)(SC[tid * 512 + c] * inv);
  }
  __syncthreads();
  for (int t = w; t < 16; t += 8) {                   // 4 Mtiles x 4 Ntiles, K=512
    const int mt = t >> 2, nt = t & 3;
    v8f acc = vzero8();
    for (int k = 0; k < cG; k += 32)
      acc = wmma_bf16(load_a16(PB, 1024, mt * 16, k, lane),
                      load_b16(VT + ((size_t)b * cE + h * cHD) * cG, cG, nt * 16, k, lane), acc);
    const int d = nt * 16 + (lane & 15);
#pragma unroll
    for (int i = 0; i < 8; ++i) {
      const int m = mt * 16 + ((lane >> 4) << 3) + i;
      OBF[(size_t)(qrow0 + m) * cE + h * cHD + d] = (bf16)acc[i];
    }
  }
}

// ---------------------------------------------------------------- host driver
extern "C" void kernel_launch(void* const* d_in, const int* in_sizes, int n_in,
                              void* d_out, int out_size, void* d_ws, size_t ws_size,
                              hipStream_t stream) {
  (void)in_sizes; (void)n_in; (void)out_size; (void)ws_size;
  // ---- inputs (setup_inputs dict order, nested dicts flattened)
  const float* points = (const float*)d_in[0];
  const float* w1  = (const float*)d_in[1];  const float* b1  = (const float*)d_in[2];
  const float* g1  = (const float*)d_in[3];  const float* bt1 = (const float*)d_in[4];
  const float* w2  = (const float*)d_in[5];  const float* b2  = (const float*)d_in[6];
  const float* g2  = (const float*)d_in[7];  const float* bt2 = (const float*)d_in[8];
  const float* w3  = (const float*)d_in[9];  const float* b3  = (const float*)d_in[10];
  const float* w4  = (const float*)d_in[11]; const float* b4  = (const float*)d_in[12];
  const float* g4  = (const float*)d_in[13]; const float* bt4 = (const float*)d_in[14];
  const float* w5  = (const float*)d_in[15]; const float* b5  = (const float*)d_in[16];
  const float* ln1g = (const float*)d_in[17]; const float* ln1b = (const float*)d_in[18];
  const float* qkvw = (const float*)d_in[19]; const float* qkvb = (const float*)d_in[20];
  const float* projw = (const float*)d_in[21]; const float* projb = (const float*)d_in[22];
  const float* ln2g = (const float*)d_in[23]; const float* ln2b = (const float*)d_in[24];
  const float* m1w = (const float*)d_in[25]; const float* m1bias = (const float*)d_in[26];
  const float* m2w = (const float*)d_in[27]; const float* m2bias = (const float*)d_in[28];

  // ---- workspace bump allocator
  char* wsp = (char*)d_ws;
  auto alloc = [&](size_t bytes) -> void* {
    void* p = (void*)wsp;
    wsp += (bytes + 255) & ~(size_t)255;
    return p;
  };
  int*   fps_idx   = (int*)alloc((size_t)BG * 4);
  float* center    = (float*)alloc((size_t)BG * 3 * 4);
  int*   group_idx = (int*)alloc((size_t)BG * cS * 4);
  int*   order     = (int*)alloc((size_t)BG * 4);
  float* cord      = (float*)alloc((size_t)BG * 3 * 4);
  float* neigh     = (float*)alloc((size_t)BG * cS * 3 * 4);
  bf16* w2b   = (bf16*)alloc((size_t)512 * 256 * 2);
  bf16* w3b   = (bf16*)alloc((size_t)1024 * 512 * 2);
  bf16* w4b   = (bf16*)alloc((size_t)2048 * 2048 * 2);
  bf16* w5b   = (bf16*)alloc((size_t)384 * 2048 * 2);
  bf16* qkvwb = (bf16*)alloc((size_t)cDEPTH * 1152 * cE * 2);
  bf16* projwb= (bf16*)alloc((size_t)cDEPTH * cE * cE * 2);
  bf16* m1wb  = (bf16*)alloc((size_t)cDEPTH * cHM * cE * 2);
  bf16* m2wb  = (bf16*)alloc((size_t)cDEPTH * cE * cHM * 2);
  bf16* Y     = (bf16*)alloc((size_t)BG * cE * 2);
  bf16* QKV   = (bf16*)alloc((size_t)BG * 1152 * 2);
  bf16* VT    = (bf16*)alloc((size_t)cB * cE * cG * 2);
  bf16* OBF   = (bf16*)alloc((size_t)BG * cE * 2);
  bf16* U     = (bf16*)alloc((size_t)BG * cHM * 2);
  float* X = (float*)d_out;   // residual stream lives directly in d_out

  auto conv = [&](const float* s, bf16* d, size_t n) {
    k_f2bf<<<(unsigned)((n + 255) / 256), 256, 0, stream>>>(s, d, (int)n);
  };
  conv(w2, w2b, (size_t)512 * 256);
  conv(w3, w3b, (size_t)1024 * 512);
  conv(w4, w4b, (size_t)2048 * 2048);
  conv(w5, w5b, (size_t)384 * 2048);
  conv(qkvw, qkvwb, (size_t)cDEPTH * 1152 * cE);
  conv(projw, projwb, (size_t)cDEPTH * cE * cE);
  conv(m1w, m1wb, (size_t)cDEPTH * cHM * cE);
  conv(m2w, m2wb, (size_t)cDEPTH * cE * cHM);

  k_fps<<<cB, 256, 0, stream>>>(points, fps_idx);
  k_knn<<<BG, 256, 0, stream>>>(points, fps_idx, group_idx, center);
  k_order<<<cB, cG, 0, stream>>>(center, order);
  k_gather<<<BG, 96, 0, stream>>>(points, center, group_idx, order, neigh, cord);
  k_enc<<<BG, 256, ENC_LDS_BYTES, stream>>>(neigh, cord, w1, b1, g1, bt1, b2, g2, bt2,
                                            b3, b4, g4, bt4, b5, w2b, w3b, w4b, w5b, X);

  for (int d = 0; d < cDEPTH; ++d) {
    k_ln<<<BG, 128, 0, stream>>>(X, ln1g + d * cE, ln1b + d * cE, Y);
    k_gemm<3><<<1152, 256, 0, stream>>>(Y, qkvwb + (size_t)d * 1152 * cE,
                                        qkvb + d * 1152, nullptr, QKV, VT, cE, 72);
    k_attn<<<cB * cHEADS * 8, 256, ATT_LDS_BYTES, stream>>>(QKV, VT, OBF);
    k_gemm<1><<<384, 256, 0, stream>>>(OBF, projwb + (size_t)d * cE * cE,
                                       projb + d * cE, X, nullptr, nullptr, cE, 24);
    k_ln<<<BG, 128, 0, stream>>>(X, ln2g + d * cE, ln2b + d * cE, Y);
    k_gemm<2><<<1536, 256, 0, stream>>>(Y, m1wb + (size_t)d * cHM * cE,
                                        m1bias + d * cHM, nullptr, U, nullptr, cE, 96);
    k_gemm<1><<<384, 256, 0, stream>>>(U, m2wb + (size_t)d * cE * cHM,
                                       m2bias + d * cE, X, nullptr, nullptr, cHM, 24);
  }
}
